// ACDA_70841190580484
// MI455X (gfx1250) — matrix operand order repaired and checked by reading, not compiled, not run on hardware
//
#include <hip/hip_runtime.h>
#include <stdint.h>

typedef __attribute__((ext_vector_type(16))) _Float16     v16h;
typedef __attribute__((ext_vector_type(8)))  float        v8f;
typedef __attribute__((ext_vector_type(4)))  float        v4f;
typedef __attribute__((ext_vector_type(4)))  unsigned int v4u;
typedef __attribute__((ext_vector_type(8)))  int          v8i;
typedef __attribute__((ext_vector_type(4)))  int          v4i;

#define C_CH    64
#define NT      36           // 576/16 N-tiles
#define PIX     32           // pixels per workgroup
#define THREADS 64           // 2 waves of 32
#define TILE_W  34           // PIX + 2 halo cols
#define XSTRIDE 34           // dense: matches TDM contiguous LDS packing
#define FB_O    144          // outputs per chunk (9 tiles = 16 channels)

// ---------------------------------------------------------------------------
// Pre-pack W_gen (576x64 f32) into per-lane B-fragment layout, split f16.
// Fragment (t,kt): lane l holds N = 16t+(l&15), K = 32kt + (l>>4)*16 + i.
// ---------------------------------------------------------------------------
__global__ void ACDA_prepack_w(const float* __restrict__ Wg,
                               _Float16* __restrict__ Whi,
                               _Float16* __restrict__ Wlo) {
    int idx = blockIdx.x * blockDim.x + threadIdx.x;
    if (idx >= NT * 2 * 32) return;
    int lane  = idx & 31;
    int kt    = (idx >> 5) & 1;
    int t     = idx >> 6;
    int o     = t * 16 + (lane & 15);
    int cbase = kt * 32 + (lane >> 4) * 16;
#pragma unroll
    for (int i = 0; i < 16; ++i) {
        float w = Wg[o * C_CH + cbase + i];
        _Float16 hi = (_Float16)w;
        Whi[idx * 16 + i] = hi;
        Wlo[idx * 16 + i] = (_Float16)(w - (float)hi);
    }
}

// ---------------------------------------------------------------------------
// Fused filter-gen GEMM (split-f16 WMMA) + unfold dot product.
// Block = (b, h, 32-pixel strip), 2 waves x 16 pixels.
// 4 chunks of 9 N-tiles (= 16 channels each): GEMM -> wave-private LDS
// f-buffer -> atomic-free reduction -> direct global store.
// ---------------------------------------------------------------------------
__global__ __launch_bounds__(THREADS) void ACDA_main(
    const float* __restrict__ x,
    const _Float16* __restrict__ Whi,
    const _Float16* __restrict__ Wlo,
    const float* __restrict__ bias,
    float* __restrict__ out)
{
    __shared__ float xs[C_CH * 3 * XSTRIDE];   // 26112 B, dense [c][r][j]
    __shared__ float fbuf[2][FB_O * 16];       // 18432 B, per-wave [o_local][p]

    const int blk   = blockIdx.x;
    const int b     = blk >> 9;                // 512 blocks per batch image
    const int rem   = blk & 511;
    const int h     = rem >> 2;
    const int wbase = (rem & 3) * PIX;
    const int tid   = threadIdx.x;
    const int lane  = tid & 31;
    const int wave  = tid >> 5;
    const size_t planeHW = 16384;              // 128*128

    // ---- stage x halo tile: 64 ch x rows h-1..h+1 x cols wbase-1..wbase+32
    if (h > 0 && h < 127 && wbase > 0) {
        // Interior: Tensor Data Mover, one 3D DMA per wave (32 channels).
        // Tile: X=34 cols (stride 128), Y=3 rows, Z=32 channels (stride 16384).
        // Right-edge (wbase==96) zero-fill via tensor_dim0 = 128 - x0.
        const int x0 = wbase - 1, y0 = h - 1;
        unsigned long long ga = (unsigned long long)(uintptr_t)
            (x + (size_t)(b * C_CH + wave * 32) * planeHW + (size_t)y0 * 128 + x0);
        unsigned int lds = (unsigned int)(uintptr_t)&xs[0]
                         + (unsigned int)(wave * 32 * 3 * XSTRIDE * 4);
        const unsigned int dim0 = 128u - (unsigned int)x0;   // remaining cols
        const unsigned int dim1 = 128u - (unsigned int)y0;   // remaining rows
        v4u g0 = { 1u,                                   // count=1, load, user
                   lds,                                  // lds_addr [63:32]
                   (unsigned int)ga,                     // global_addr lo
                   ((unsigned int)(ga >> 32) & 0x01FFFFFFu) | (2u << 30) }; // type=2
        v8i g1 = { (int)(2u << 16),                      // data_size=4B
                   (int)((dim0 & 0xFFFFu) << 16),        // tensor_dim0 lo16 @48
                   (int)((dim0 >> 16) | ((dim1 & 0xFFFFu) << 16)),
                   (int)((dim1 >> 16) | (34u << 16)),    // tile_dim0=34 @112
                   (int)(3u | (32u << 16)),              // tile_dim1=3, tile_dim2=32
                   (int)128,                             // tensor_dim0_stride lo32
                   (int)((16384u & 0xFFFFu) << 16),      // stride0 hi=0 | stride1 lo16
                   (int)(16384u >> 16) };                // stride1 hi32
        v4i g2 = { 32, 0, 0, 0 };                        // tensor_dim2=32
        v4i g3 = { 0, 0, 0, 0 };
#if defined(__clang_major__) && __clang_major__ >= 23
        v8i g4 = { 0, 0, 0, 0, 0, 0, 0, 0 };
        __builtin_amdgcn_tensor_load_to_lds(g0, g1, g2, g3, g4, 0);
#else
        __builtin_amdgcn_tensor_load_to_lds(g0, g1, g2, g3, 0);
#endif
        __builtin_amdgcn_s_wait_tensorcnt(0);
    } else {
        // Boundary: scalar staging with zero halo (dense layout -> xs[i]).
        for (int i = tid; i < C_CH * 3 * TILE_W; i += THREADS) {
            int c  = i / (3 * TILE_W);
            int r  = (i / TILE_W) % 3;
            int j  = i % TILE_W;
            int gh = h - 1 + r;
            int gw = wbase - 1 + j;
            float xv = 0.0f;
            if (gh >= 0 && gh < 128 && gw >= 0 && gw < 128)
                xv = x[(size_t)(b * C_CH + c) * planeHW + (size_t)gh * 128 + gw];
            xs[i] = xv;
        }
    }
    __syncthreads();

    const int pixbase = wave * 16;
    const int mrow    = lane & 15;     // A: M / D: N index
    const int half    = lane >> 4;
    const int koff    = half * 8;
    float* fb = fbuf[wave];

    // ---- split-f16 A fragments (built once, reused for all 36 N-tiles)
    v16h a_hi[2], a_lo[2];
#pragma unroll
    for (int kt = 0; kt < 2; ++kt) {
#pragma unroll
        for (int i = 0; i < 16; ++i) {
            int K = half * 16 + i;
            int c = kt * 32 + K;
            float xv = xs[(c * 3 + 1) * XSTRIDE + (pixbase + mrow) + 1];
            _Float16 hi = (_Float16)xv;
            a_hi[kt][i] = hi;
            a_lo[kt][i] = (_Float16)(xv - (float)hi);
        }
    }

    for (int chunk = 0; chunk < 4; ++chunk) {
        // ---- GEMM: 9 tiles -> bias+ReLU -> wave-private f-buffer
        for (int tt = 0; tt < 9; ++tt) {
            int t = chunk * 9 + tt;
            v8f acc = {};
#pragma unroll
            for (int kt = 0; kt < 2; ++kt) {
                int fidx = ((t * 2 + kt) * 32 + lane) * 16;
                v16h bh = *(const v16h*)(Whi + fidx);
                v16h bl = *(const v16h*)(Wlo + fidx);
                // hi*hi + lo*hi + hi*lo (drop lo*lo): ~fp32 accuracy
                acc = __builtin_amdgcn_wmma_f32_16x16x32_f16(false, a_hi[kt], false, bh,
                                                             (short)0, acc, false, false);
                acc = __builtin_amdgcn_wmma_f32_16x16x32_f16(false, a_lo[kt], false, bh,
                                                             (short)0, acc, false, false);
                acc = __builtin_amdgcn_wmma_f32_16x16x32_f16(false, a_hi[kt], false, bl,
                                                             (short)0, acc, false, false);
            }
            float bo = bias[t * 16 + mrow];
            // D: lane holds N = o (fixed), M = v + 8*half -> consecutive pixels
            float* dst = fb + (tt * 16 + mrow) * 16 + 8 * half;
            v4f f0, f1;
#pragma unroll
            for (int v = 0; v < 4; ++v) {
                float f = acc[v] + bo;     f0[v] = f > 0.0f ? f : 0.0f;
                float g = acc[v + 4] + bo; f1[v] = g > 0.0f ? g : 0.0f;
            }
            *(v4f*)dst       = f0;        // two ds_store_b128
            *((v4f*)dst + 1) = f1;
        }
        // ---- reduction: 16 channels x 16 pixels, atomic-free, wave-private
#pragma unroll
        for (int it = 0; it < 8; ++it) {
            int idx = lane + 32 * it;          // 0..255
            int cl  = idx >> 4;                // channel within chunk
            int pr  = idx & 15;                // pixel within wave
            int c   = chunk * 16 + cl;
            const float* fp   = fb + (cl * 9) * 16 + pr;
            const float* prow = &xs[c * 3 * XSTRIDE + pixbase + pr];
            float s = 0.0f;
#pragma unroll
            for (int j = 0; j < 9; ++j)        // j/3, j%3 fold to constants
                s += fp[j * 16] * prow[(j / 3) * XSTRIDE + (j % 3)];
            out[(size_t)(b * C_CH + c) * planeHW + (size_t)h * 128
                + (wbase + pixbase + pr)] = s;
        }
    }
}

extern "C" void kernel_launch(void* const* d_in, const int* in_sizes, int n_in,
                              void* d_out, int out_size, void* d_ws, size_t ws_size,
                              hipStream_t stream) {
    (void)in_sizes; (void)n_in; (void)out_size; (void)ws_size;
    const float* x  = (const float*)d_in[0];
    const float* Wg = (const float*)d_in[1];
    const float* bg = (const float*)d_in[2];
    float* out      = (float*)d_out;

    _Float16* Whi = (_Float16*)d_ws;                 // 36864 halves
    _Float16* Wlo = Whi + (size_t)NT * 2 * 32 * 16;  // 36864 halves

    int frag_lanes = NT * 2 * 32;                    // 2304
    ACDA_prepack_w<<<(frag_lanes + 127) / 128, 128, 0, stream>>>(Wg, Whi, Wlo);

    // B * H * (W/PIX) = 8 * 128 * 4 = 4096 blocks
    ACDA_main<<<4096, THREADS, 0, stream>>>(x, Whi, Wlo, bg, out);
}
